// MultiHeadedAttention_25786983645493
// MI455X (gfx1250) — matrix operand order repaired and checked by reading, not compiled, not run on hardware
//
#include <hip/hip_runtime.h>

// ---------------------------------------------------------------------------
// MultiHeadedAttention forward on gfx1250 (MI455X), wave32 + WMMA bf16
// + async global->LDS tile staging (ASYNCcnt) with double-buffered GEMM.
// B=2, S=2048, D_MODEL=1024, H=16, DK=64.
// ---------------------------------------------------------------------------

#define DM   1024
#define NH   16
#define DKH  64
#define BB   2
#define SEQ  2048
#define MTOT (BB * SEQ)   // 4096 rows

typedef __attribute__((ext_vector_type(16))) __bf16 v16bf;
typedef __attribute__((ext_vector_type(8)))  float  v8f;

#if __has_builtin(__builtin_amdgcn_global_load_async_to_lds_b128)
#define HAVE_ASYNC_LDS 1
#else
#define HAVE_ASYNC_LDS 0
#endif

// exact parameter pointee type of the async builtin (from hipcc diagnostic):
// __attribute__((vector_size(16))) int, in AS(1) for global / AS(3) for LDS.
typedef int async_v4i __attribute__((vector_size(16)));

// 16-byte global -> LDS copy; async (no VGPR data path, ASYNCcnt) when available.
__device__ __forceinline__ void copy_b128_g2l(const __bf16* g, __bf16* l) {
#if HAVE_ASYNC_LDS
    __builtin_amdgcn_global_load_async_to_lds_b128(
        (__attribute__((address_space(1))) async_v4i*)g,
        (__attribute__((address_space(3))) async_v4i*)l,
        /*offset=*/0, /*cpol=*/0);
#else
    *reinterpret_cast<uint4*>(l) = *reinterpret_cast<const uint4*>(g);
#endif
}

__device__ __forceinline__ void wait_async_lds() {
#if HAVE_ASYNC_LDS
#if __has_builtin(__builtin_amdgcn_s_wait_asynccnt)
    __builtin_amdgcn_s_wait_asynccnt(0);
#else
    asm volatile("s_wait_asynccnt 0x0" ::: "memory");
#endif
#endif
}

__device__ __forceinline__ v8f wmma_bf16(v16bf a, v16bf b, v8f c) {
    // D = A(16x32 bf16) * B(32x16 bf16) + C(16x16 f32)
    return __builtin_amdgcn_wmma_f32_16x16x32_bf16(
        /*neg_a=*/false, a, /*neg_b=*/false, b,
        /*c_mod=*/(short)0, c, /*reuse_a=*/false, /*reuse_b=*/false);
}

// Build a v16bf fragment from two 16-byte chunks (p0: elems 0..7, p1: elems 8..15)
__device__ __forceinline__ v16bf load_frag_2x8(const __bf16* p0, const __bf16* p1) {
    v16bf r;
    reinterpret_cast<uint4*>(&r)[0] = *reinterpret_cast<const uint4*>(p0);
    reinterpret_cast<uint4*>(&r)[1] = *reinterpret_cast<const uint4*>(p1);
    return r;
}

// ---------------------------------------------------------------------------
// fp32 -> bf16 conversion (RNE via compiler cast), 4 elems/thread
// ---------------------------------------------------------------------------
__global__ void cvt_f32_to_bf16_kernel(const float* __restrict__ in,
                                       __bf16* __restrict__ out, int n) {
    int i = (blockIdx.x * blockDim.x + threadIdx.x) * 4;
    if (i >= n) return;
    float4 v = *reinterpret_cast<const float4*>(in + i);
    alignas(8) __bf16 t[4] = { (__bf16)v.x, (__bf16)v.y, (__bf16)v.z, (__bf16)v.w };
    *reinterpret_cast<uint2*>(out + i) = *reinterpret_cast<const uint2*>(t);
}

// ---------------------------------------------------------------------------
// Y[M,N] = X[M,K](bf16) @ W[N,K]^T(bf16) + bias[N](f32)
// OUT_F32 ? f32 output : bf16 output.
// Block tile 128x128, K-step 64, 4 waves (128 thr), each wave a 64x64 tile.
// Double-buffered LDS (2 x 16KB per operand = 64KB) filled via async copies:
// next K-slice is issued before computing the current one.
// LDS row stride = 64 elems (128B, 16B multiple -> ds_load_b128 stays legal).
// ---------------------------------------------------------------------------
template<int OUT_F32>
__global__ __launch_bounds__(128)
void gemm_nt_bf16_kernel(const __bf16* __restrict__ A, const __bf16* __restrict__ Bm,
                         const float* __restrict__ bias, void* __restrict__ Y,
                         int M, int N, int K) {
    __shared__ __bf16 As[2][128 * 64];   // 2 x 16 KB
    __shared__ __bf16 Bs[2][128 * 64];   // 2 x 16 KB

    const int tid  = threadIdx.x;
    const int wid  = tid >> 5;
    const int lane = tid & 31;
    const int half = lane >> 4;
    const int ln   = lane & 15;
    const int wm   = (wid >> 1) * 64;      // wave row offset in block tile
    const int wn   = (wid & 1)  * 64;      // wave col offset in block tile
    const int m0   = blockIdx.y * 128;
    const int n0   = blockIdx.x * 128;

    v8f acc[4][4];
    const v8f vzero = {0.f,0.f,0.f,0.f,0.f,0.f,0.f,0.f};
#pragma unroll
    for (int i = 0; i < 4; ++i)
#pragma unroll
        for (int j = 0; j < 4; ++j) acc[i][j] = vzero;

    // thread t stages one 64-elem row of each operand tile (8 x b128 each)
    auto issue_tile = [&](int k0, int buf) {
        const __bf16* ga = &A [(size_t)(m0 + tid) * K + k0];
        const __bf16* gb = &Bm[(size_t)(n0 + tid) * K + k0];
        __bf16* la = &As[buf][tid * 64];
        __bf16* lb = &Bs[buf][tid * 64];
#pragma unroll
        for (int j = 0; j < 8; ++j) {
            copy_b128_g2l(ga + j * 8, la + j * 8);
            copy_b128_g2l(gb + j * 8, lb + j * 8);
        }
    };

    const int NIT = K / 64;
    issue_tile(0, 0);
    wait_async_lds();
    __syncthreads();

    for (int it = 0; it < NIT; ++it) {
        const int buf = it & 1;
        if (it + 1 < NIT) issue_tile((it + 1) * 64, buf ^ 1);   // prefetch next slice

#pragma unroll
        for (int ks = 0; ks < 2; ++ks) {
            v16bf af[4], bf[4];
#pragma unroll
            for (int f = 0; f < 4; ++f) {
                // A frag: row m = wm+f*16+ln ; k chunks at 32*ks + 8*half and +16
                const __bf16* pa = &As[buf][(wm + f * 16 + ln) * 64 + ks * 32 + half * 8];
                af[f] = load_frag_2x8(pa, pa + 16);
                // B frag: col n = wn+f*16+ln ; k chunks at 32*ks + 8*half and +16
                const __bf16* pb = &Bs[buf][(wn + f * 16 + ln) * 64 + ks * 32 + half * 8];
                bf[f] = load_frag_2x8(pb, pb + 16);
            }
#pragma unroll
            for (int i = 0; i < 4; ++i)
#pragma unroll
                for (int j = 0; j < 4; ++j)
                    acc[i][j] = wmma_bf16(af[i], bf[j], acc[i][j]);
        }
        wait_async_lds();       // next slice resident in LDS
        __syncthreads();        // all waves done with current slice + see new one
    }

    // epilogue: C/D layout -> row r+8*half, col ln
#pragma unroll
    for (int i = 0; i < 4; ++i)
#pragma unroll
        for (int j = 0; j < 4; ++j)
#pragma unroll
            for (int r = 0; r < 8; ++r) {
                int m = m0 + wm + i * 16 + r + half * 8;
                int n = n0 + wn + j * 16 + ln;
                float v = acc[i][j][r] + bias[n];
                if (OUT_F32)
                    reinterpret_cast<float*>(Y)[(size_t)m * N + n] = v;
                else
                    reinterpret_cast<__bf16*>(Y)[(size_t)m * N + n] = (__bf16)v;
            }
}

// ---------------------------------------------------------------------------
// Flash attention: O[b,s,h*64+d] = softmax(Q K^T / 8, mask) V  (bf16 in/out)
// Grid: (SEQ/64 q-tiles, B*H). 128 threads = 4 waves; each wave owns 16 q-rows.
// K tile staged via async global->LDS; V tile transposed through VGPRs.
// ---------------------------------------------------------------------------
__global__ __launch_bounds__(128)
void attention_flash_kernel(const __bf16* __restrict__ Q, const __bf16* __restrict__ Kp,
                            const __bf16* __restrict__ Vp,
                            const unsigned char* __restrict__ mask,
                            __bf16* __restrict__ O) {
    __shared__ __bf16 Ks [64 * 72];        // K tile   [key][d]   (padded rows, 144B)
    __shared__ __bf16 VTs[64 * 72];        // V tile transposed [d][key]
    __shared__ __bf16 Ps [4][16 * 72];     // per-wave P staging [q_local][key_local]

    const int tid  = threadIdx.x;
    const int wid  = tid >> 5;
    const int lane = tid & 31;
    const int half = lane >> 4;
    const int ln   = lane & 15;
    const int bh = blockIdx.y;
    const int b  = bh >> 4;
    const int h  = bh & 15;
    const int qw = blockIdx.x * 64 + wid * 16;           // wave q base
    const size_t base = (size_t)b * SEQ * DM + (size_t)h * DKH;

    // persistent Q A-fragments (k-dim = head dim, 2 steps of 32)
    v16bf aq[2];
    {
        const __bf16* qp = &Q[base + (size_t)(qw + ln) * DM];
#pragma unroll
        for (int ks = 0; ks < 2; ++ks) {
            const __bf16* p = qp + ks * 32 + half * 8;
            aq[ks] = load_frag_2x8(p, p + 16);
        }
    }

    float mi[8], li[8];
    v8f o[4];
    const v8f vzero = {0.f,0.f,0.f,0.f,0.f,0.f,0.f,0.f};
#pragma unroll
    for (int r = 0; r < 8; ++r) { mi[r] = -3.0e38f; li[r] = 0.f; }
#pragma unroll
    for (int f = 0; f < 4; ++f) o[f] = vzero;

    const int ldrow = tid >> 1;            // 0..63 (key row)
    const int ldcol = (tid & 1) * 32;      // 0 or 32 (d half)

    for (int t = 0; t < SEQ / 64; ++t) {
        const int kt = t * 64;
        // --- K tile via async global->LDS (ASYNCcnt) ---
        {
            const __bf16* g = &Kp[base + (size_t)(kt + ldrow) * DM + ldcol];
            __bf16* l = &Ks[ldrow * 72 + ldcol];
#pragma unroll
            for (int j = 0; j < 4; ++j) copy_b128_g2l(g + j * 8, l + j * 8);
        }
        // --- V tile, transposed through VGPRs into VTs[d][key] ---
        {
            const uint4* g = reinterpret_cast<const uint4*>(
                &Vp[base + (size_t)(kt + ldrow) * DM + ldcol]);
            alignas(16) __bf16 e[32];
            uint4* ev = reinterpret_cast<uint4*>(e);
#pragma unroll
            for (int j = 0; j < 4; ++j) ev[j] = g[j];
#pragma unroll
            for (int j = 0; j < 32; ++j) VTs[(ldcol + j) * 72 + ldrow] = e[j];
        }
        wait_async_lds();
        __syncthreads();

        // --- scores S = Q K^T (16 x 64) ---
        v8f s[4];
#pragma unroll
        for (int f = 0; f < 4; ++f) s[f] = vzero;
#pragma unroll
        for (int ks = 0; ks < 2; ++ks) {
#pragma unroll
            for (int f = 0; f < 4; ++f) {
                // B frag: n = key = f*16+ln ; k = d chunks at 32*ks + 8*half and +16
                const __bf16* pb = &Ks[(f * 16 + ln) * 72 + ks * 32 + half * 8];
                v16bf bk = load_frag_2x8(pb, pb + 16);
                s[f] = wmma_bf16(aq[ks], bk, s[f]);
            }
        }
        // --- scale + mask ---
#pragma unroll
        for (int f = 0; f < 4; ++f)
#pragma unroll
            for (int r = 0; r < 8; ++r) {
                int qg = qw + r + half * 8;
                int kg = kt + f * 16 + ln;
                float v = s[f][r] * 0.125f;
                unsigned char mb = mask[((size_t)b * SEQ + qg) * SEQ + kg];
                s[f][r] = mb ? v : -1.0e9f;
            }
        // --- online softmax (rows live in 16-lane groups) ---
#pragma unroll
        for (int r = 0; r < 8; ++r) {
            float rm = fmaxf(fmaxf(s[0][r], s[1][r]), fmaxf(s[2][r], s[3][r]));
#pragma unroll
            for (int off = 1; off < 16; off <<= 1) rm = fmaxf(rm, __shfl_xor(rm, off));
            float mnew = fmaxf(mi[r], rm);
            float rs = 0.f;
#pragma unroll
            for (int f = 0; f < 4; ++f) {
                float p = __expf(s[f][r] - mnew);
                s[f][r] = p;
                rs += p;
            }
#pragma unroll
            for (int off = 1; off < 16; off <<= 1) rs += __shfl_xor(rs, off);
            float alpha = __expf(mi[r] - mnew);
            li[r] = li[r] * alpha + rs;
            mi[r] = mnew;
#pragma unroll
            for (int f = 0; f < 4; ++f) o[f][r] *= alpha;
        }
        // --- restage P (C layout -> A layout) via per-wave LDS ---
#pragma unroll
        for (int f = 0; f < 4; ++f)
#pragma unroll
            for (int r = 0; r < 8; ++r)
                Ps[wid][(r + half * 8) * 72 + f * 16 + ln] = (__bf16)s[f][r];
        // --- O += P V ---
#pragma unroll
        for (int ks = 0; ks < 2; ++ks) {
            const __bf16* pp = &Ps[wid][ln * 72 + ks * 32 + half * 8];
            v16bf pa = load_frag_2x8(pp, pp + 16);
#pragma unroll
            for (int f = 0; f < 4; ++f) {
                // B frag: n = d = f*16+ln ; k = key chunks at 32*ks + 8*half and +16
                const __bf16* pv = &VTs[(f * 16 + ln) * 72 + ks * 32 + half * 8];
                v16bf bv = load_frag_2x8(pv, pv + 16);
                o[f] = wmma_bf16(pa, bv, o[f]);
            }
        }
        __syncthreads();
    }

    // --- normalize + store bf16 ---
#pragma unroll
    for (int f = 0; f < 4; ++f)
#pragma unroll
        for (int r = 0; r < 8; ++r) {
            int qg = qw + r + half * 8;
            float v = o[f][r] / li[r];
            O[base + (size_t)qg * DM + f * 16 + ln] = (__bf16)v;
        }
}

// ---------------------------------------------------------------------------
// Host side
// ---------------------------------------------------------------------------
extern "C" void kernel_launch(void* const* d_in, const int* in_sizes, int n_in,
                              void* d_out, int out_size, void* d_ws, size_t ws_size,
                              hipStream_t stream) {
    const float* q    = (const float*)d_in[0];
    const float* k    = (const float*)d_in[1];
    const float* v    = (const float*)d_in[2];
    const unsigned char* mask = (const unsigned char*)d_in[3];
    const float* wq_w = (const float*)d_in[4];
    const float* wq_b = (const float*)d_in[5];
    const float* wk_w = (const float*)d_in[6];
    const float* wk_b = (const float*)d_in[7];
    const float* wv_w = (const float*)d_in[8];
    const float* wv_b = (const float*)d_in[9];
    const float* wo_w = (const float*)d_in[10];
    const float* wo_b = (const float*)d_in[11];

    char* ws = (char*)d_ws;
    size_t off = 0;
    auto alloc_bf16 = [&](size_t elems) -> __bf16* {
        __bf16* p = (__bf16*)(ws + off);
        off += (elems * sizeof(__bf16) + 255) & ~(size_t)255;
        return p;
    };
    __bf16* Xbuf = alloc_bf16((size_t)MTOT * DM);  // activation staging (8 MB), also attn out
    __bf16* Wbuf = alloc_bf16((size_t)DM * DM);    // weight staging (2 MB)
    __bf16* Qp   = alloc_bf16((size_t)MTOT * DM);  // projected Q (8 MB)
    __bf16* Kp   = alloc_bf16((size_t)MTOT * DM);  // projected K
    __bf16* Vp   = alloc_bf16((size_t)MTOT * DM);  // projected V
    (void)ws_size; (void)in_sizes; (void)n_in; (void)out_size;

    const int nAct = MTOT * DM;     // 4 Mi elems
    const int nWgt = DM * DM;       // 1 Mi elems
    dim3 cvtBlk(256);
    dim3 cvtActGrid(nAct / 1024);
    dim3 cvtWgtGrid(nWgt / 1024);
    dim3 gemmGrid(DM / 128, MTOT / 128);   // (8, 32)
    dim3 gemmBlk(128);
    dim3 attnGrid(SEQ / 64, BB * NH);      // (32, 32)
    dim3 attnBlk(128);

    // Q projection
    cvt_f32_to_bf16_kernel<<<cvtActGrid, cvtBlk, 0, stream>>>(q, Xbuf, nAct);
    cvt_f32_to_bf16_kernel<<<cvtWgtGrid, cvtBlk, 0, stream>>>(wq_w, Wbuf, nWgt);
    gemm_nt_bf16_kernel<0><<<gemmGrid, gemmBlk, 0, stream>>>(Xbuf, Wbuf, wq_b, Qp, MTOT, DM, DM);
    // K projection
    cvt_f32_to_bf16_kernel<<<cvtActGrid, cvtBlk, 0, stream>>>(k, Xbuf, nAct);
    cvt_f32_to_bf16_kernel<<<cvtWgtGrid, cvtBlk, 0, stream>>>(wk_w, Wbuf, nWgt);
    gemm_nt_bf16_kernel<0><<<gemmGrid, gemmBlk, 0, stream>>>(Xbuf, Wbuf, wk_b, Kp, MTOT, DM, DM);
    // V projection
    cvt_f32_to_bf16_kernel<<<cvtActGrid, cvtBlk, 0, stream>>>(v, Xbuf, nAct);
    cvt_f32_to_bf16_kernel<<<cvtWgtGrid, cvtBlk, 0, stream>>>(wv_w, Wbuf, nWgt);
    gemm_nt_bf16_kernel<0><<<gemmGrid, gemmBlk, 0, stream>>>(Xbuf, Wbuf, wv_b, Vp, MTOT, DM, DM);
    // attention (output reuses Xbuf)
    attention_flash_kernel<<<attnGrid, attnBlk, 0, stream>>>(Qp, Kp, Vp, mask, Xbuf);
    // output projection -> f32 d_out
    cvt_f32_to_bf16_kernel<<<cvtWgtGrid, cvtBlk, 0, stream>>>(wo_w, Wbuf, nWgt);
    gemm_nt_bf16_kernel<1><<<gemmGrid, gemmBlk, 0, stream>>>(Xbuf, Wbuf, wo_b, d_out, MTOT, DM, DM);
}
